// _FFTConv_66365834658073
// MI455X (gfx1250) — matrix operand order repaired and checked by reading, not compiled, not run on hardware
//
#include <hip/hip_runtime.h>

// ---- CDNA5 (gfx1250, wave32) vector types -------------------------------
typedef __bf16        v16bf __attribute__((ext_vector_type(16)));
typedef float         v8f   __attribute__((ext_vector_type(8)));
typedef unsigned int  v8u   __attribute__((ext_vector_type(8)));
typedef unsigned int  v4u   __attribute__((ext_vector_type(4)));
typedef int           v8i   __attribute__((ext_vector_type(8)));

#define CIN    64
#define COUT   64
#define KH     31
#define KW     31
#define IMG_H  224
#define IMG_W  224
#define NB     16
#define PAD    15
#define TILE_W 32
#define TILE_H 4
#define WTILES (IMG_W / TILE_W)     // 7
#define HTILES (IMG_H / TILE_H)     // 56
#define KSLICES (CIN * KH)          // 1984 K-steps of 32 (31 taps + 1 zero)
#define SLICE_BYTES 4096            // one (c,u) slice: 64 out-ch x 32 k, bf16
#define IN_ROWS (TILE_H + KH - 1)   // 34 staged input rows per channel

// LDS byte offset of a __shared__ object (generic -> AS3 -> int)
typedef __attribute__((address_space(3))) const void lds_cvoid;
__device__ __forceinline__ unsigned lds_off(const void* p) {
    return (unsigned)(unsigned long long)(lds_cvoid*)p;
}

// ---------------------------------------------------------------------------
// Kernel 1: repack weights fp32 [o][c][u][v] -> bf16 slices, each slice laid
// out as the exact B-fragment register image: slice ks = (c*31+u), within it
// byte offset = ntile*1024 + lane*32 + j*4 + p*2 holds w[o=16*ntile+(lane&15)]
// [c][u][k] with k = (lane<16?0:16) + 2j + p, k==31 zero-padded.
// Total: 1984*4096 B = 7.75 MiB in d_ws (L2-resident on 192MB L2).
// ---------------------------------------------------------------------------
__global__ __launch_bounds__(256) void convert_weights(
        const float* __restrict__ wgt, __bf16* __restrict__ w2) {
    int idx = blockIdx.x * 256 + threadIdx.x;
    const int total = KSLICES * 2048;
    if (idx >= total) return;
    int ks = idx >> 11;            // slice
    int r  = idx & 2047;           // bf16 element within slice
    int t  = r >> 9;               // ntile (16 output channels each)
    int l  = (r >> 4) & 31;        // lane
    int j  = (r >> 1) & 7;         // B VGPR index (packs k=kb+2j, kb+2j+1)
    int p  = r & 1;
    int n  = l & 15;
    int kb = (l < 16) ? 0 : 16;
    int k  = kb + 2 * j + p;
    int o  = 16 * t + n;
    int c  = ks / KH;
    int u  = ks - c * KH;
    float v = 0.0f;
    if (k < KW) v = wgt[(((o * CIN) + c) * KH + u) * KW + k];
    w2[idx] = (__bf16)v;
}

// ---------------------------------------------------------------------------
// TDM: DMA one 4KB weight slice global->LDS (D# groups 0/1, 1-D tile,
// data_size=8B, tile_dim0=512). Issued by one wave; tracked on TENSORcnt.
// ---------------------------------------------------------------------------
__device__ __forceinline__ void tdm_load_slice(const __bf16* w2, int ks,
                                               unsigned lds_dst) {
    unsigned long long ga =
        (unsigned long long)w2 + (unsigned long long)ks * SLICE_BYTES;
    v4u g0;
    g0.x = 1u;                                  // count=1, user descriptor
    g0.y = lds_dst;                             // lds_addr[31:0]
    g0.z = (unsigned)ga;                        // global_addr[31:0]
    g0.w = (unsigned)(ga >> 32) | (2u << 30);   // global_addr[56:32] | type=2
    v8i g1;
    g1[0] = (3 << 16);            // workgroup_mask=0 | data_size=3 (8B units)
    g1[1] = (int)(512u << 16);    // tensor_dim0[15:0]=512 (in bits 63:48)
    g1[2] = (int)(1u << 16);      // tensor_dim0[31:16]=0 | tensor_dim1[15:0]=1
    g1[3] = (int)(512u << 16);    // tensor_dim1[31:16]=0 | tile_dim0=512
    g1[4] = 0;                    // tile_dim1=0, tile_dim2=0 (1-D tile)
    g1[5] = 512;                  // tensor_dim0_stride[31:0]
    g1[6] = 0;
    g1[7] = 0;
    asm volatile("tensor_load_to_lds %0, %1" :: "s"(g0), "s"(g1) : "memory");
}

// ---------------------------------------------------------------------------
// Kernel 2: implicit-GEMM conv. 256 threads = 8 waves.
// Block tile: 128 output pixels (32 wide x 4 rows) x 64 output channels.
// Wave (mpair,npair) owns a 2x2 grid of 16x16 C tiles -> 4 WMMAs per K-step.
// A: dword-aligned Toeplitz slices via dual-alignment LDS row copies.
// B: TDM-double-buffered pre-swizzled slices; 2 aligned 16B LDS loads/tile.
// ---------------------------------------------------------------------------
__global__ __launch_bounds__(256) void fftconv_wmma(
        const float* __restrict__ sig, const __bf16* __restrict__ w2,
        const float* __restrict__ bias, float* __restrict__ out) {

    __shared__ __bf16 inA[IN_ROWS][64];              // row copy, even phase
    __shared__ __bf16 inB[IN_ROWS][64];              // row copy shifted by 1
    __shared__ alignas(32) unsigned wbuf[2][SLICE_BYTES / 4];  // double buffer

    const int blk = blockIdx.x;
    const int wt  = blk % WTILES;
    const int th  = (blk / WTILES) % HTILES;
    const int b   = blk / (WTILES * HTILES);
    const int h0  = th * TILE_H;
    const int w0  = wt * TILE_W;

    const int tid   = threadIdx.x;
    const int lane  = tid & 31;
    const int wv    = tid >> 5;
    const int mpair = wv & 3;     // image row within 4-row tile
    const int npair = wv >> 2;    // which 32-channel half

    // A fragment addressing (16-bit A 16x32; contiguous k order used here, same as B):
    // lanes 0-15: M=lane, K={0..15}; lanes 16-31: M=lane-16, K={16..31}
    const int mloc = lane & 15;
    const int akb  = (lane < 16) ? 0 : 16;
    const int nloc = lane & 15;

    const unsigned wbuf_base = lds_off(&wbuf[0][0]);

    v8f acc00 = {}, acc01 = {}, acc10 = {}, acc11 = {};

    if (tid < 32) tdm_load_slice(w2, 0, wbuf_base);   // prologue DMA

    for (int c = 0; c < CIN; ++c) {
        for (int u = 0; u < KH; ++u) {
            const int ks = c * KH + u;
            if (tid < 32) __builtin_amdgcn_s_wait_tensorcnt(0); // cur slice done
            __syncthreads();            // publish DMA; all waves past prev step
            const int nxt = ks + 1;
            if (tid < 32 && nxt < KSLICES)                       // prefetch next
                tdm_load_slice(w2, nxt,
                               wbuf_base + (unsigned)(nxt & 1) * SLICE_BYTES);

            if (u == 0) {
                // Stage padded input patch for channel c, dual-alignment copies
                for (int i = tid; i < IN_ROWS * 64; i += 256) {
                    int r  = i >> 6, j = i & 63;
                    int hh = h0 - PAD + r;
                    int ww = w0 - PAD + j;
                    float v = 0.0f;
                    if ((unsigned)hh < IMG_H && (unsigned)ww < IMG_W)
                        v = sig[(((b * CIN) + c) * IMG_H + hh) * IMG_W + ww];
                    __bf16 bv = (__bf16)v;
                    inA[r][j] = bv;
                    if (j > 0) inB[r][j - 1] = bv;
                }
                __syncthreads();
            }

            // ---- A fragments: Toeplitz row slices, all loads dword-aligned
            const int rrow = mpair + u;
            v16bf a0, a1;
            {
                const int base = mloc + akb;                    // pixel half 0
                const unsigned* rp = (base & 1)
                    ? (const unsigned*)&inB[rrow][base - 1]
                    : (const unsigned*)&inA[rrow][base];
                v8u tv;
#pragma unroll
                for (int q = 0; q < 4; ++q) { tv[q] = rp[q]; tv[4 + q] = rp[4 + q]; }
                a0 = __builtin_bit_cast(v16bf, tv);
            }
            {
                const int base = 16 + mloc + akb;               // pixel half 1
                const unsigned* rp = (base & 1)
                    ? (const unsigned*)&inB[rrow][base - 1]
                    : (const unsigned*)&inA[rrow][base];
                v8u tv;
#pragma unroll
                for (int q = 0; q < 4; ++q) { tv[q] = rp[q]; tv[4 + q] = rp[4 + q]; }
                a1 = __builtin_bit_cast(v16bf, tv);
            }

            // ---- B fragments: pre-swizzled, 32 contiguous bytes per lane
            const unsigned* wcur = &wbuf[ks & 1][0];
            v16bf b0 = *(const v16bf*)(wcur + (npair * 2 + 0) * 256 + lane * 8);
            v16bf b1 = *(const v16bf*)(wcur + (npair * 2 + 1) * 256 + lane * 8);

            acc00 = __builtin_amdgcn_wmma_f32_16x16x32_bf16(
                        false, a0, false, b0, (short)0, acc00, false, false);
            acc01 = __builtin_amdgcn_wmma_f32_16x16x32_bf16(
                        false, a0, false, b1, (short)0, acc01, false, false);
            acc10 = __builtin_amdgcn_wmma_f32_16x16x32_bf16(
                        false, a1, false, b0, (short)0, acc10, false, false);
            acc11 = __builtin_amdgcn_wmma_f32_16x16x32_bf16(
                        false, a1, false, b1, (short)0, acc11, false, false);
        }
    }

    // ---- Epilogue: C/D layout -> VGPR i holds M=i (lanes 0-15) / i+8 (16-31)
    const int   hrow = h0 + mpair;
    const int   o0   = 16 * (npair * 2 + 0) + nloc;
    const int   o1   = 16 * (npair * 2 + 1) + nloc;
    const float bv0  = bias[o0];
    const float bv1  = bias[o1];
    const size_t base0 = (((size_t)(b * COUT) + o0) * IMG_H + hrow) * IMG_W + w0;
    const size_t base1 = (((size_t)(b * COUT) + o1) * IMG_H + hrow) * IMG_W + w0;
#pragma unroll
    for (int i = 0; i < 8; ++i) {
        int m = (lane < 16) ? i : (i + 8);
        out[base0 + m]      = acc00[i] + bv0;
        out[base0 + 16 + m] = acc10[i] + bv0;
        out[base1 + m]      = acc01[i] + bv1;
        out[base1 + 16 + m] = acc11[i] + bv1;
    }
}

// ---------------------------------------------------------------------------
extern "C" void kernel_launch(void* const* d_in, const int* in_sizes, int n_in,
                              void* d_out, int out_size, void* d_ws, size_t ws_size,
                              hipStream_t stream) {
    const float* sig  = (const float*)d_in[0];   // (16,64,224,224) f32
    const float* wgt  = (const float*)d_in[1];   // (64,64,31,31)   f32
    const float* bias = (const float*)d_in[2];   // (64,)           f32
    float* out = (float*)d_out;                  // (16,64,224,224) f32
    __bf16* w2 = (__bf16*)d_ws;                  // 7.75 MiB repacked weights

    {
        const int total  = KSLICES * 2048;
        const int blocks = (total + 255) / 256;
        convert_weights<<<blocks, 256, 0, stream>>>(wgt, w2);
    }
    {
        const int blocks = NB * HTILES * WTILES;  // 6272
        fftconv_wmma<<<blocks, 256, 0, stream>>>(sig, w2, bias, out);
    }
}
